// HMM_20736102105797
// MI455X (gfx1250) — matrix operand order repaired and checked by reading, not compiled, not run on hardware
//
#include <hip/hip_runtime.h>

#define NSTEP  16384
#define NSTATE 512

// ---- CDNA5 WMMA vector types ----
typedef __attribute__((ext_vector_type(16))) __bf16    v16bf;
typedef __attribute__((ext_vector_type(8)))  float     v8f;
typedef __attribute__((ext_vector_type(4)))  unsigned  u32x4;

union TileVec { u32x4 q[2]; v16bf v; };   // 32 bytes, 16-byte aligned

// bf16 round-to-nearest-even helpers (bit-level, matches hardware RNE)
__device__ __forceinline__ unsigned short f2bf(float f) {
  unsigned u = __float_as_uint(f);
  unsigned r = u + 0x7FFFu + ((u >> 16) & 1u);
  return (unsigned short)(r >> 16);
}
__device__ __forceinline__ float bf2f(unsigned short h) {
  return __uint_as_float(((unsigned)h) << 16);
}

// =====================================================================
// Kernel 1: pre-tile A (forward) and A^T (backward) into a contiguous-k
// v_wmma_f32_16x16x32_bf16 B-operand layout, split into bf16 hi + lo.
// B-tile (32x16): lane l holds column n=l%16; l<16 -> K 0..15 (2/VGPR),
// l>=16 -> K 16..31. Tile = 256 dwords, dword index = l*8 + v.
// ws dword layout: [F_hi 128K][F_lo 128K][B_hi 128K][B_lo 128K][betas]
// =====================================================================
__global__ void hmm_prep_tiles(const float* __restrict__ A,
                               unsigned* __restrict__ ws) {
  int id = blockIdx.x * blockDim.x + threadIdx.x;   // 0 .. 262143
  int m    = id >> 17;                              // 0 = A, 1 = A^T
  int r    = id & 0x1FFFF;
  int tile = r >> 8;                                // nt*16 + kt
  int d    = r & 255;
  int l = d >> 3, v = d & 7;
  int nt = tile >> 4, kt = tile & 15;
  int n = nt * 16 + (l & 15);
  int k = kt * 32 + ((l >= 16) ? 16 : 0) + 2 * v;   // element pair (k,k+1)
  float x0, x1;
  if (m == 0) { x0 = A[k * NSTATE + n];  x1 = A[(k + 1) * NSTATE + n]; }
  else        { x0 = A[n * NSTATE + k];  x1 = A[n * NSTATE + k + 1]; }
  unsigned short h0 = f2bf(x0), h1 = f2bf(x1);
  float r0 = x0 - bf2f(h0), r1 = x1 - bf2f(h1);
  unsigned* hi = ws + m * 262144;
  unsigned* lo = hi + 131072;
  hi[tile * 256 + d] = (unsigned)h0        | ((unsigned)h1        << 16);
  lo[tile * 256 + d] = (unsigned)f2bf(r0)  | ((unsigned)f2bf(r1)  << 16);
}

// =====================================================================
// Fused WMMA matvec for both of a wave's output tiles:
//   y0[n] , y1[n]  =  (a @ M)[nt0*16+lane] , (a @ M)[nt1*16+lane]
// The packed bf16 vector is BROADCAST into all 16 rows of the WMMA
// A-operand: in the contiguous k order used here, row M=l lives in lane l
// (K 0-15) and lane l+16 (K 16-31), so every lane loads
// base = kt*16 + ((lane&16)>>1). Lanes 0-15 share one LDS address,
// lanes 16-31 another -> bank broadcast, no EXEC masking, no zeroing.
// D rows 1..15 are redundant copies of y; we only read acc[0].
// bf16 hi/lo split on vector and matrix -> ~fp32 accuracy, 3 products.
// Two accumulators interleaved: independent WMMAs issue back-to-back.
// =====================================================================
__device__ __forceinline__ void matvec_pair(const unsigned* __restrict__ tHi,
                                            const unsigned* __restrict__ tLo,
                                            const unsigned* aHi,
                                            const unsigned* aLo,
                                            int nt0, int nt1, int lane,
                                            float& y0, float& y1) {
  v8f acc0 = {0.f, 0.f, 0.f, 0.f, 0.f, 0.f, 0.f, 0.f};
  v8f acc1 = {0.f, 0.f, 0.f, 0.f, 0.f, 0.f, 0.f, 0.f};
  const int aofs = (lane & 16) >> 1;    // 0 for lanes 0-15, 8 for 16-31
#pragma unroll 4
  for (int kt = 0; kt < 16; ++kt) {
    const int base = kt * 16 + aofs;
    TileVec aH, aL;
    aH.q[0] = *(const u32x4*)(aHi + base);
    aH.q[1] = *(const u32x4*)(aHi + base + 4);
    aL.q[0] = *(const u32x4*)(aLo + base);
    aL.q[1] = *(const u32x4*)(aLo + base + 4);
    const u32x4* bh0 = (const u32x4*)(tHi + (((nt0 << 4) + kt) << 8));
    const u32x4* bl0 = (const u32x4*)(tLo + (((nt0 << 4) + kt) << 8));
    const u32x4* bh1 = (const u32x4*)(tHi + (((nt1 << 4) + kt) << 8));
    const u32x4* bl1 = (const u32x4*)(tLo + (((nt1 << 4) + kt) << 8));
    // warm the near cache one tile ahead (A is L2-resident)
    __builtin_prefetch((const void*)(bh0 + 64 + lane * 2), 0, 3);
    __builtin_prefetch((const void*)(bh1 + 64 + lane * 2), 0, 3);
    TileVec bH0, bL0, bH1, bL1;
    bH0.q[0] = bh0[lane * 2]; bH0.q[1] = bh0[lane * 2 + 1];
    bL0.q[0] = bl0[lane * 2]; bL0.q[1] = bl0[lane * 2 + 1];
    bH1.q[0] = bh1[lane * 2]; bH1.q[1] = bh1[lane * 2 + 1];
    bL1.q[0] = bl1[lane * 2]; bL1.q[1] = bl1[lane * 2 + 1];
    acc0 = __builtin_amdgcn_wmma_f32_16x16x32_bf16(false, aH.v, false, bH0.v,
                                                   (short)0, acc0, false, false);
    acc1 = __builtin_amdgcn_wmma_f32_16x16x32_bf16(false, aH.v, false, bH1.v,
                                                   (short)0, acc1, false, false);
    acc0 = __builtin_amdgcn_wmma_f32_16x16x32_bf16(false, aL.v, false, bH0.v,
                                                   (short)0, acc0, false, false);
    acc1 = __builtin_amdgcn_wmma_f32_16x16x32_bf16(false, aL.v, false, bH1.v,
                                                   (short)0, acc1, false, false);
    acc0 = __builtin_amdgcn_wmma_f32_16x16x32_bf16(false, aH.v, false, bL0.v,
                                                   (short)0, acc0, false, false);
    acc1 = __builtin_amdgcn_wmma_f32_16x16x32_bf16(false, aH.v, false, bL1.v,
                                                   (short)0, acc1, false, false);
  }
  y0 = acc0[0];   // VGPR0, lanes 0..15 = (M=0, N=lane)
  y1 = acc1[0];
}

// Block-wide sum; the single __syncthreads per step lives here and also
// orders the LDS pack-writes (double-buffered) for the next iteration.
__device__ __forceinline__ float block_sum(float v, int lane, int wave,
                                           float* wsum) {
  float s = v;
#pragma unroll
  for (int off = 16; off >= 1; off >>= 1) s += __shfl_xor(s, off, 32);
  if (lane == 0) wsum[wave] = s;
  __syncthreads();
  float tot = 0.f;
#pragma unroll
  for (int i = 0; i < 16; ++i) tot += wsum[i];
  return tot;
}

// Pack vector into LDS as bf16 hi/lo pairs (A-operand feed).
__device__ __forceinline__ void pack_pair(float v, int lane, int nt,
                                          unsigned* pkHi, unsigned* pkLo) {
  float vn = __shfl_down(v, 1, 32);
  if ((lane < 16) && ((lane & 1) == 0)) {
    unsigned short h0 = f2bf(v), h1 = f2bf(vn);
    float r0 = v - bf2f(h0), r1 = vn - bf2f(h1);
    pkHi[nt * 8 + (lane >> 1)] = (unsigned)h0       | ((unsigned)h1       << 16);
    pkLo[nt * 8 + (lane >> 1)] = (unsigned)f2bf(r0) | ((unsigned)f2bf(r1) << 16);
  }
}

// =====================================================================
// Kernel 2: both serial chains, concurrently on two WGPs.
// Block 0: forward (alphas -> d_out). Block 1: backward (betas -> ws).
// Each chain self-normalizes; scalings cancel exactly in the gammas.
// One barrier per step: LDS vector double-buffered; packed copy uses
// the previous step's 1/sum (scale-invariant; bounded drift).
// =====================================================================
__global__ void __launch_bounds__(512, 1)
hmm_chains(const float* __restrict__ obs, const float* __restrict__ pi0,
           const unsigned* __restrict__ ws, float* __restrict__ alphas,
           float* __restrict__ betas) {
  __shared__ __align__(16) unsigned apkHi[512];   // 2 banks of 256 dwords
  __shared__ __align__(16) unsigned apkLo[512];
  __shared__ float wsum[16];

  const int tid  = threadIdx.x;
  const int lane = tid & 31;
  const int wave = tid >> 5;
  const int nt0 = wave * 2, nt1 = nt0 + 1;
  const int n0 = nt0 * 16 + lane;   // valid when lane < 16
  const int n1 = nt1 * 16 + lane;
  const bool act = lane < 16;

  if (blockIdx.x == 0) {            // ----------- forward -----------
    const unsigned* tHi = ws;
    const unsigned* tLo = ws + 131072;
    float v0 = act ? pi0[n0] * obs[n0] : 0.f;    // t = 0
    float v1 = act ? pi0[n1] * obs[n1] : 0.f;
    float inv = 1.0f / block_sum(v0 + v1, lane, wave, wsum);
    v0 *= inv; v1 *= inv;
    if (act) { alphas[n0] = v0; alphas[n1] = v1; }
    pack_pair(v0, lane, nt0, apkHi, apkLo);      // bank 0
    pack_pair(v1, lane, nt1, apkHi, apkLo);
    __syncthreads();
    float invc = inv;                            // stale-scale carry
    int p = 0;
    for (int t = 1; t < NSTEP; ++t) {
      const unsigned *rdH = apkHi + (p << 8),  *rdL = apkLo + (p << 8);
      unsigned       *wrH = apkHi + ((p ^ 1) << 8), *wrL = apkLo + ((p ^ 1) << 8);
      float y0, y1;
      matvec_pair(tHi, tLo, rdH, rdL, nt0, nt1, lane, y0, y1);
      const float* ob = obs + t * NSTATE;
      __builtin_prefetch((const void*)(ob + NSTATE + lane), 0, 3);
      float w0 = act ? y0 * ob[n0] : 0.f;
      float w1 = act ? y1 * ob[n1] : 0.f;
      pack_pair(w0 * invc, lane, nt0, wrH, wrL); // pre-barrier pack
      pack_pair(w1 * invc, lane, nt1, wrH, wrL);
      float s = 1.0f / block_sum(w0 + w1, lane, wave, wsum);  // barrier
      if (act) { alphas[t * NSTATE + n0] = w0 * s;            // exact norm
                 alphas[t * NSTATE + n1] = w1 * s; }
      invc = s;                                  // carry current 1/sum
      p ^= 1;
    }
  } else {                          // ----------- backward -----------
    const unsigned* tHi = ws + 262144;            // A^T tiles
    const unsigned* tLo = ws + 393216;
    const float* obN = obs + (NSTEP - 1) * NSTATE;
    float b0 = act ? 1.0f : 0.f, b1 = b0;
    float inv = 1.0f / block_sum(b0 + b1, lane, wave, wsum);   // = 1/512
    b0 *= inv; b1 *= inv;
    if (act) {
      betas[(NSTEP - 1) * NSTATE + n0] = b0;
      betas[(NSTEP - 1) * NSTATE + n1] = b1;
    }
    float w0 = act ? b0 * obN[n0] : 0.f;          // w = b_{t+1} o obs_{t+1}
    float w1 = act ? b1 * obN[n1] : 0.f;
    pack_pair(w0, lane, nt0, apkHi, apkLo);       // bank 0
    pack_pair(w1, lane, nt1, apkHi, apkLo);
    __syncthreads();
    float invc = inv;
    int p = 0;
    for (int t = NSTEP - 2; t >= 0; --t) {
      const unsigned *rdH = apkHi + (p << 8),  *rdL = apkLo + (p << 8);
      unsigned       *wrH = apkHi + ((p ^ 1) << 8), *wrL = apkLo + ((p ^ 1) << 8);
      float y0, y1;
      matvec_pair(tHi, tLo, rdH, rdL, nt0, nt1, lane, y0, y1); // w @ A^T
      const float* ob = obs + t * NSTATE;
      __builtin_prefetch((const void*)(ob - NSTATE + lane), 0, 3);
      // next step's operand: (b_t o obs_t), scale-free (stale 1/sum)
      float u0 = act ? y0 * invc * ob[n0] : 0.f;
      float u1 = act ? y1 * invc * ob[n1] : 0.f;
      pack_pair(u0, lane, nt0, wrH, wrL);
      pack_pair(u1, lane, nt1, wrH, wrL);
      float s = 1.0f / block_sum(act ? (y0 + y1) : 0.f, lane, wave, wsum);
      if (act) { betas[t * NSTATE + n0] = y0 * s;              // exact norm
                 betas[t * NSTATE + n1] = y1 * s; }
      invc = s;
      p ^= 1;
    }
  }
}

// =====================================================================
// Kernel 3: gammas = (alphas o betas) / rowsum, in place over d_out.
// =====================================================================
__global__ void hmm_gamma(const float* __restrict__ betas,
                          float* __restrict__ g) {
  __shared__ float psum[8];
  int t = blockIdx.x, tid = threadIdx.x;   // 256 threads, 2 cols each
  int i0 = t * NSTATE + tid, i1 = i0 + 256;
  float p0 = g[i0] * betas[i0];
  float p1 = g[i1] * betas[i1];
  float s = p0 + p1;
#pragma unroll
  for (int off = 16; off >= 1; off >>= 1) s += __shfl_xor(s, off, 32);
  if ((tid & 31) == 0) psum[tid >> 5] = s;
  __syncthreads();
  float tot = 0.f;
#pragma unroll
  for (int i = 0; i < 8; ++i) tot += psum[i];
  g[i0] = p0 / tot;
  g[i1] = p1 / tot;
}

extern "C" void kernel_launch(void* const* d_in, const int* in_sizes, int n_in,
                              void* d_out, int out_size, void* d_ws,
                              size_t ws_size, hipStream_t stream) {
  (void)in_sizes; (void)n_in; (void)out_size; (void)ws_size;
  const float* obs = (const float*)d_in[0];
  const float* A   = (const float*)d_in[1];
  const float* pi0 = (const float*)d_in[2];
  float*    out   = (float*)d_out;
  unsigned* ws    = (unsigned*)d_ws;
  float*    betas = (float*)(ws + 524288);   // after 2 MB of bf16 tiles

  hmm_prep_tiles<<<1024, 256, 0, stream>>>(A, ws);
  hmm_chains<<<2, 512, 0, stream>>>(obs, pi0, ws, out, betas);
  hmm_gamma<<<NSTEP, 256, 0, stream>>>(betas, out);
}
